// TreeIsomorphismNetwork_2937757630885
// MI455X (gfx1250) — matrix operand order repaired and checked by reading, not compiled, not run on hardware
//
#include <hip/hip_runtime.h>
#include <hip/hip_bf16.h>

// ---------------------------------------------------------------------------
// TreeIsomorphismNetwork for MI455X (gfx1250, wave32, WMMA)
// B=64, N0=4096, N1=256, D_IN=HID=256, OUT=141
// ---------------------------------------------------------------------------

typedef __bf16 bf16_t;
typedef __attribute__((ext_vector_type(16))) __bf16 bf16x16;
typedef __attribute__((ext_vector_type(8)))  float  f32x8;

#define TIN_B    64
#define TIN_N0   4096
#define TIN_N1   256
#define TIN_D    256
#define TIN_OUT  141
#define TIN_EPS  1e-5f

// ---------------------------------------------------------------------------
// Kernel 1: deterministic scatter-sum pool (children -> parents) + total sum.
// grid = (64 batches, 4 feature chunks of 64), block = 64 threads.
// Each thread owns one feature column of the 256-parent LDS accumulator,
// so no atomics are needed and summation order is fixed (deterministic).
// ---------------------------------------------------------------------------
__global__ void __launch_bounds__(64)
pool1_kernel(const float* __restrict__ h0, const int* __restrict__ parent,
             float* __restrict__ p1, float* __restrict__ sumh0) {
  __shared__ float acc[TIN_N1 * 64];  // 64 KB LDS
  const int b  = blockIdx.x;
  const int c0 = blockIdx.y * 64;
  const int t  = threadIdx.x;

  for (int p = 0; p < TIN_N1; ++p) acc[p * 64 + t] = 0.0f;

  const float* src = h0 + (size_t)b * TIN_N0 * TIN_D + c0 + t;
#pragma unroll 4
  for (int n = 0; n < TIN_N0; ++n) {
    int   par = parent[n];
    float v   = src[(size_t)n * TIN_D];
    acc[par * 64 + t] += v;
  }

  float tot = 0.0f;
  for (int p = 0; p < TIN_N1; ++p) {
    float v = acc[p * 64 + t];
    tot += v;
    p1[((size_t)b * TIN_N1 + p) * TIN_D + c0 + t] = v;
  }
  sumh0[b * TIN_D + c0 + t] = tot;
}

// ---------------------------------------------------------------------------
// Kernel 2: weight prep — transpose 256x256 f32 W[k][n] into bf16 Wt[n][k]
// so WMMA B fragments become contiguous 32B loads.
// ---------------------------------------------------------------------------
__global__ void __launch_bounds__(256)
transpose_to_bf16(const float* __restrict__ W, bf16_t* __restrict__ Wt) {
  const int n = blockIdx.x;    // output row
  const int k = threadIdx.x;   // output col (coalesced write)
  Wt[(size_t)n * TIN_D + k] = (bf16_t)W[(size_t)k * TIN_D + n];
}

// ---------------------------------------------------------------------------
// Kernel 3: GEMM  out[M,256] = A[M,256] (f32, cvt->bf16) @ Wt^T + bias
// Wt is bf16 [N=256][K=256] row-major (pre-transposed weights).
// Block = 256 threads = 8 waves; wave -> 32x32 tile (4 WMMA accumulators);
// block tile = 64(M) x 128(N); grid = (M/64, 2). K loop: 8 x v_wmma bf16.
// ---------------------------------------------------------------------------
__global__ void __launch_bounds__(256)
gemm256_bf16(const float* __restrict__ A, const bf16_t* __restrict__ Wt,
             const float* __restrict__ bias, float* __restrict__ out, int M) {
  constexpr int K = TIN_D, N = TIN_D;
  const int lane = threadIdx.x & 31;
  const int wave = threadIdx.x >> 5;
  const int wm   = wave & 1;   // 0..1 along M
  const int wn   = wave >> 1;  // 0..3 along N
  const int mBase = blockIdx.x * 64 + wm * 32;
  const int nBase = blockIdx.y * 128 + wn * 32;

  // A-fragment addressing (16x32 bf16 tile, contiguous K order matching B):
  // lanes 0-15: row m, elems 0..15 = K 0..15
  // lanes16-31: row m, elems 0..15 = K 16..31
  const int mA0 = mBase + (lane & 15);
  const int mA1 = mA0 + 16;
  const int aK  = (lane >> 4) * 16;
  // B-fragment: lane n holds contiguous K span of 16 (lo lanes K0..15, hi K16..31)
  const int nB  = lane & 15;
  const int bK  = (lane >> 4) * 16;

  f32x8 c00 = {}, c01 = {}, c10 = {}, c11 = {};

  for (int k0 = 0; k0 < K; k0 += 32) {
    bf16x16 a0, a1;
    const float* r0 = A + (size_t)mA0 * K + k0 + aK;
    const float* r1 = A + (size_t)mA1 * K + k0 + aK;
#pragma unroll
    for (int i = 0; i < 8; ++i) {
      a0[i]     = (bf16_t)r0[i];
      a0[i + 8] = (bf16_t)r0[i + 8];
      a1[i]     = (bf16_t)r1[i];
      a1[i + 8] = (bf16_t)r1[i + 8];
    }
    bf16x16 b0 = *(const bf16x16*)(Wt + (size_t)(nBase + nB) * K + k0 + bK);
    bf16x16 b1 = *(const bf16x16*)(Wt + (size_t)(nBase + 16 + nB) * K + k0 + bK);

    c00 = __builtin_amdgcn_wmma_f32_16x16x32_bf16(false, a0, false, b0, (short)0, c00, false, false);
    c01 = __builtin_amdgcn_wmma_f32_16x16x32_bf16(false, a0, false, b1, (short)0, c01, false, false);
    c10 = __builtin_amdgcn_wmma_f32_16x16x32_bf16(false, a1, false, b0, (short)0, c10, false, false);
    c11 = __builtin_amdgcn_wmma_f32_16x16x32_bf16(false, a1, false, b1, (short)0, c11, false, false);
  }

  // C/D layout: elem r of lane L -> row = r + (L>=16 ? 8 : 0), col = L&15
  const int col0  = nBase + (lane & 15);
  const int col1  = col0 + 16;
  const int rbase = (lane >> 4) * 8;
  const float bv0 = bias[col0];
  const float bv1 = bias[col1];
#pragma unroll
  for (int r = 0; r < 8; ++r) {
    const int row0 = mBase + rbase + r;
    const int row1 = row0 + 16;
    out[(size_t)row0 * N + col0] = c00[r] + bv0;
    out[(size_t)row0 * N + col1] = c01[r] + bv1;
    out[(size_t)row1 * N + col0] = c10[r] + bv0;
    out[(size_t)row1 * N + col1] = c11[r] + bv1;
  }
}

// ---------------------------------------------------------------------------
// BatchNorm (training-mode batch stats), deterministic two-stage reduction.
// ---------------------------------------------------------------------------
__global__ void __launch_bounds__(256)
bn_stats_partial(const float* __restrict__ Y, float* __restrict__ psum,
                 float* __restrict__ psq, int M, int G) {
  const int g = blockIdx.x, c = threadIdx.x;
  const int rpb = M / G;
  float s = 0.0f, q = 0.0f;
  const float* p = Y + (size_t)g * rpb * TIN_D + c;
  for (int r = 0; r < rpb; ++r) { float v = p[(size_t)r * TIN_D]; s += v; q += v * v; }
  psum[g * TIN_D + c] = s;
  psq [g * TIN_D + c] = q;
}

__global__ void __launch_bounds__(256)
bn_finalize(const float* __restrict__ psum, const float* __restrict__ psq,
            int G, int M, const float* __restrict__ gamma,
            const float* __restrict__ beta, float* __restrict__ scale,
            float* __restrict__ shift) {
  const int c = threadIdx.x;
  float s = 0.0f, q = 0.0f;
  for (int g = 0; g < G; ++g) { s += psum[g * TIN_D + c]; q += psq[g * TIN_D + c]; }
  const float inv_m = 1.0f / (float)M;
  const float mean  = s * inv_m;
  const float var   = q * inv_m - mean * mean;
  const float sc    = gamma[c] * rsqrtf(var + TIN_EPS);
  scale[c] = sc;
  shift[c] = beta[c] - mean * sc;
}

__global__ void __launch_bounds__(256)
bn_apply_relu(const float* __restrict__ Y, const float* __restrict__ scale,
              const float* __restrict__ shift, float* __restrict__ out) {
  const size_t i = (size_t)blockIdx.x * TIN_D + threadIdx.x;  // grid.x == M rows
  const int    c = threadIdx.x;
  const float  v = Y[i] * scale[c] + shift[c];
  out[i] = v > 0.0f ? v : 0.0f;
}

// ---------------------------------------------------------------------------
// Root pool: parent2 is all-zero with N2=1, so p2[b] = sum_n h1[b,n,:].
// This is identical to sum(h1, axis=1) needed for the logits.
// ---------------------------------------------------------------------------
__global__ void __launch_bounds__(256)
pool2_kernel(const float* __restrict__ h1, float* __restrict__ p2) {
  const int b = blockIdx.x, d = threadIdx.x;
  const float* p = h1 + (size_t)b * TIN_N1 * TIN_D + d;
  float s = 0.0f;
  for (int n = 0; n < TIN_N1; ++n) s += p[(size_t)n * TIN_D];
  p2[b * TIN_D + d] = s;
}

// ---------------------------------------------------------------------------
// Final logits: three tiny 64x141x256 GEMVs fused. grid=64, block=160.
// ---------------------------------------------------------------------------
__global__ void __launch_bounds__(160)
final_logits(const float* __restrict__ sumh0, const float* __restrict__ sumh1,
             const float* __restrict__ h2,
             const float* __restrict__ Wp0, const float* __restrict__ bp0,
             const float* __restrict__ Wp1, const float* __restrict__ bp1,
             const float* __restrict__ Wp2, const float* __restrict__ bp2,
             float* __restrict__ out) {
  const int b = blockIdx.x;
  const int o = threadIdx.x;
  if (o >= TIN_OUT) return;
  float acc = bp0[o] + bp1[o] + bp2[o];
  const float* x0 = sumh0 + b * TIN_D;
  const float* x1 = sumh1 + b * TIN_D;
  const float* x2 = h2    + b * TIN_D;
  for (int k = 0; k < TIN_D; ++k) {
    acc += x0[k] * Wp0[k * TIN_OUT + o];
    acc += x1[k] * Wp1[k * TIN_OUT + o];
    acc += x2[k] * Wp2[k * TIN_OUT + o];
  }
  out[b * TIN_OUT + o] = acc;
}

// ---------------------------------------------------------------------------
// Launch sequence
// ---------------------------------------------------------------------------
extern "C" void kernel_launch(void* const* d_in, const int* in_sizes, int n_in,
                              void* d_out, int out_size, void* d_ws, size_t ws_size,
                              hipStream_t stream) {
  const float* h0    = (const float*)d_in[0];
  const int*   par1  = (const int*)d_in[1];
  // d_in[2] = parent2 (all zeros; N2=1 root pool handled analytically)
  const float* W1_1  = (const float*)d_in[3];
  const float* b1_1  = (const float*)d_in[4];
  const float* g1_1  = (const float*)d_in[5];
  const float* be1_1 = (const float*)d_in[6];
  const float* W2_1  = (const float*)d_in[7];
  const float* b2_1  = (const float*)d_in[8];
  const float* gbn1  = (const float*)d_in[9];
  const float* bbn1  = (const float*)d_in[10];
  const float* W1_2  = (const float*)d_in[11];
  const float* b1_2  = (const float*)d_in[12];
  const float* g1_2  = (const float*)d_in[13];
  const float* be1_2 = (const float*)d_in[14];
  const float* W2_2  = (const float*)d_in[15];
  const float* b2_2  = (const float*)d_in[16];
  const float* gbn2  = (const float*)d_in[17];
  const float* bbn2  = (const float*)d_in[18];
  const float* Wp0   = (const float*)d_in[19];
  const float* bp0   = (const float*)d_in[20];
  const float* Wp1   = (const float*)d_in[21];
  const float* bp1   = (const float*)d_in[22];
  const float* Wp2   = (const float*)d_in[23];
  const float* bp2   = (const float*)d_in[24];
  float* out = (float*)d_out;

  // Workspace layout (~34.5 MB total)
  char* ws = (char*)d_ws;
  const size_t BIG = (size_t)16384 * TIN_D * sizeof(float);  // 16 MB
  float* buf0 = (float*)(ws);            // p1, then H, then h1
  float* buf1 = (float*)(ws + BIG);      // pre-BN Z buffers
  char*  S    = ws + 2 * BIG;
  float* sumh0 = (float*)(S);
  float* p2    = (float*)(S + 1 * 65536);   // == sum(h1, axis=1)
  float* psum  = (float*)(S + 2 * 65536);
  float* psq   = (float*)(S + 3 * 65536);
  float* scale = (float*)(S + 4 * 65536);
  float* shift = (float*)(S + 4 * 65536 + 4096);
  float* yS    = (float*)(S + 4 * 65536 + 8192);
  float* hS    = (float*)(S + 5 * 65536 + 8192);
  float* h2    = (float*)(S + 6 * 65536 + 8192);
  bf16_t* Wt11 = (bf16_t*)(S + 7 * 65536 + 8192);
  bf16_t* Wt21 = Wt11 + 65536;
  bf16_t* Wt12 = Wt11 + 2 * 65536;
  bf16_t* Wt22 = Wt11 + 3 * 65536;

  const int Mbig = TIN_B * TIN_N1;  // 16384

  // 0) weight prep: f32 -> transposed bf16 (one-time per call, 512 KB total)
  transpose_to_bf16<<<dim3(256), dim3(256), 0, stream>>>(W1_1, Wt11);
  transpose_to_bf16<<<dim3(256), dim3(256), 0, stream>>>(W2_1, Wt21);
  transpose_to_bf16<<<dim3(256), dim3(256), 0, stream>>>(W1_2, Wt12);
  transpose_to_bf16<<<dim3(256), dim3(256), 0, stream>>>(W2_2, Wt22);

  // 1) scatter pool (streams the 256 MB input once) + sum(h0, axis=1)
  pool1_kernel<<<dim3(TIN_B, 4), dim3(64), 0, stream>>>(h0, par1, buf0, sumh0);

  // 2) layer-1 MLP: Linear -> BN -> ReLU -> Linear -> BN -> ReLU
  gemm256_bf16<<<dim3(Mbig / 64, 2), dim3(256), 0, stream>>>(buf0, Wt11, b1_1, buf1, Mbig);
  bn_stats_partial<<<dim3(64), dim3(256), 0, stream>>>(buf1, psum, psq, Mbig, 64);
  bn_finalize<<<dim3(1), dim3(256), 0, stream>>>(psum, psq, 64, Mbig, g1_1, be1_1, scale, shift);
  bn_apply_relu<<<dim3(Mbig), dim3(256), 0, stream>>>(buf1, scale, shift, buf0);  // H

  gemm256_bf16<<<dim3(Mbig / 64, 2), dim3(256), 0, stream>>>(buf0, Wt21, b2_1, buf1, Mbig);
  bn_stats_partial<<<dim3(64), dim3(256), 0, stream>>>(buf1, psum, psq, Mbig, 64);
  bn_finalize<<<dim3(1), dim3(256), 0, stream>>>(psum, psq, 64, Mbig, gbn1, bbn1, scale, shift);
  bn_apply_relu<<<dim3(Mbig), dim3(256), 0, stream>>>(buf1, scale, shift, buf0);  // h1

  // 3) root pool: p2 = sum(h1, axis=1)  (also the layer-1 logit pooling term)
  pool2_kernel<<<dim3(TIN_B), dim3(256), 0, stream>>>(buf0, p2);

  // 4) layer-2 MLP on [64, 256]
  gemm256_bf16<<<dim3(1, 2), dim3(256), 0, stream>>>(p2, Wt12, b1_2, yS, TIN_B);
  bn_stats_partial<<<dim3(1), dim3(256), 0, stream>>>(yS, psum, psq, TIN_B, 1);
  bn_finalize<<<dim3(1), dim3(256), 0, stream>>>(psum, psq, 1, TIN_B, g1_2, be1_2, scale, shift);
  bn_apply_relu<<<dim3(TIN_B), dim3(256), 0, stream>>>(yS, scale, shift, hS);

  gemm256_bf16<<<dim3(1, 2), dim3(256), 0, stream>>>(hS, Wt22, b2_2, yS, TIN_B);
  bn_stats_partial<<<dim3(1), dim3(256), 0, stream>>>(yS, psum, psq, TIN_B, 1);
  bn_finalize<<<dim3(1), dim3(256), 0, stream>>>(psum, psq, 1, TIN_B, gbn2, bbn2, scale, shift);
  bn_apply_relu<<<dim3(TIN_B), dim3(256), 0, stream>>>(yS, scale, shift, h2);

  // 5) fused prediction heads
  final_logits<<<dim3(TIN_B), dim3(160), 0, stream>>>(
      sumh0, p2, h2, Wp0, bp0, Wp1, bp1, Wp2, bp2, out);

  (void)in_sizes; (void)n_in; (void)out_size; (void)ws_size;
}
